// Net_63496796504296
// MI455X (gfx1250) — matrix operand order repaired and checked by reading, not compiled, not run on hardware
//
#include <hip/hip_runtime.h>
#include <hip/hip_bf16.h>

#define NBLK 128
#define TSEQ 8192
#define HID  1024
#define NFUT 16

typedef __attribute__((ext_vector_type(16))) __bf16 v16bf;
typedef __attribute__((ext_vector_type(8)))  __bf16 v8bf;
typedef __attribute__((ext_vector_type(8)))  float  v8f;
typedef __attribute__((ext_vector_type(4)))  unsigned int u32x4;
typedef __attribute__((ext_vector_type(8)))  int i32x8;
typedef __attribute__((ext_vector_type(4)))  int i32x4;

static __device__ __forceinline__ unsigned short f32_to_bf16(float f) {
    union { float f; unsigned u; } v; v.f = f;
    unsigned r = v.u + 0x7FFFu + ((v.u >> 16) & 1u);
    return (unsigned short)(r >> 16);
}
static __device__ __forceinline__ float bf16_to_f32(unsigned short h) {
    union { unsigned u; float f; } v; v.u = ((unsigned)h) << 16; return v.f;
}
static __device__ __forceinline__ float wave_reduce(float v) {
    #pragma unroll
    for (int off = 16; off > 0; off >>= 1) v += __shfl_xor(v, off, 32);
    return v;
}
static __device__ __forceinline__ float sigm(float x) { return 1.0f / (1.0f + __expf(-x)); }
static __device__ __forceinline__ float leaky(float x) { return x >= 0.0f ? x : 0.8f * x; }

// Device-scope grid barrier for a persistent kernel (all NBLK blocks resident).
static __device__ __forceinline__ void grid_sync(unsigned* cnt, unsigned* phase,
                                                 unsigned& lp, unsigned nblk) {
    __syncthreads();
    if (threadIdx.x == 0) {
        lp += 1;
        __threadfence();
        unsigned prev = __hip_atomic_fetch_add(cnt, 1u, __ATOMIC_ACQ_REL, __HIP_MEMORY_SCOPE_AGENT);
        if (prev == nblk - 1u) {
            __hip_atomic_store(cnt, 0u, __ATOMIC_RELAXED, __HIP_MEMORY_SCOPE_AGENT);
            __hip_atomic_store(phase, lp, __ATOMIC_RELEASE, __HIP_MEMORY_SCOPE_AGENT);
        } else {
            while (__hip_atomic_load(phase, __ATOMIC_ACQUIRE, __HIP_MEMORY_SCOPE_AGENT) < lp) {
                __builtin_amdgcn_s_sleep(1);
            }
        }
    }
    __syncthreads();
}

// ---------------------------------------------------------------------------
// TDM: issue a 2-D tile load (bf16 elements) global -> LDS via one D#.
// D# packing per CDNA5 ISA section 8.3/8.4; groups 2/3 zero => 2-D tensor,
// tile_dim2 = 0 (unused). data_size = 1 (2-byte elements), type = 2 (image).
// This toolchain's builtin takes 6 operands (4th SGPR group + cpol last).
static __device__ __forceinline__ void tdm_load_tile_2d(
    const unsigned short* gsrc, unsigned lds_byte_off,
    unsigned tensor_w, unsigned tensor_h, unsigned stride_w,
    unsigned tile_w, unsigned tile_h) {
    unsigned long long ga = (unsigned long long)(uintptr_t)gsrc;
    u32x4 g0;
    g0[0] = 1u;                                             // count=1, user D#
    g0[1] = lds_byte_off;                                   // lds_addr[31:0]
    g0[2] = (unsigned)(ga & 0xFFFFFFFFu);                   // global_addr[31:0]
    g0[3] = (unsigned)((ga >> 32) & 0x01FFFFFFu) | (2u << 30); // addr[56:32] | type=2
    i32x8 g1;
    g1[0] = (int)(1u << 16);                                // data_size=2B, mask=0
    g1[1] = (int)((tensor_w & 0xFFFFu) << 16);              // tensor_dim0[15:0]
    g1[2] = (int)((tensor_w >> 16) | ((tensor_h & 0xFFFFu) << 16)); // d0 hi | d1 lo
    g1[3] = (int)((tensor_h >> 16) | (tile_w << 16));       // d1 hi | tile_dim0
    g1[4] = (int)tile_h;                                    // tile_dim1 | tile_dim2=0
    g1[5] = (int)stride_w;                                  // dim0_stride[31:0]
    g1[6] = 0;                                              // dim0_stride hi | dim1_stride lo
    g1[7] = 0;
    i32x4 gz4; gz4[0] = 0; gz4[1] = 0; gz4[2] = 0; gz4[3] = 0;
    i32x8 gz8; gz8[0] = 0; gz8[1] = 0; gz8[2] = 0; gz8[3] = 0;
    gz8[4] = 0; gz8[5] = 0; gz8[6] = 0; gz8[7] = 0;
    __builtin_amdgcn_tensor_load_to_lds(g0, g1, gz4, gz4, gz8, 0);
}

// ---------------------------------------------------------------------------
// fp32 -> bf16 weight conversion for the 4 H x H MLP matrices.
__global__ void convert_w(const float* __restrict__ a, const float* __restrict__ b,
                          const float* __restrict__ c, const float* __restrict__ d,
                          unsigned short* __restrict__ o) {
    int i = blockIdx.x * blockDim.x + threadIdx.x;
    if (i < HID * HID) {
        o[i]                 = f32_to_bf16(a[i]);
        o[1 * HID * HID + i] = f32_to_bf16(b[i]);
        o[2 * HID * HID + i] = f32_to_bf16(c[i]);
        o[3 * HID * HID + i] = f32_to_bf16(d[i]);
    }
}

// ---------------------------------------------------------------------------
// Persistent 2-layer LSTM scan. One wave per hidden unit j; each wave computes
// gate rows {j, H+j, 2H+j, 3H+j}. Weights stay L2-resident across all steps.
__global__ void lstm_scan_kernel(
    const float* __restrict__ x,
    const float* __restrict__ Wih0, const float* __restrict__ Whh0,
    const float* __restrict__ bih0, const float* __restrict__ bhh0,
    const float* __restrict__ Wih1, const float* __restrict__ Whh1,
    const float* __restrict__ bih1, const float* __restrict__ bhh1,
    float* __restrict__ h0, float* __restrict__ c0,
    float* __restrict__ h1, float* __restrict__ c1,
    unsigned short* __restrict__ hs, unsigned* __restrict__ bar) {
    const int lane = threadIdx.x & 31;
    const int wid  = threadIdx.x >> 5;
    const int j    = blockIdx.x * 8 + wid;       // 0..1023
    unsigned lp = 0;
    unsigned* cnt = bar; unsigned* phase = bar + 1;

    for (int t = 0; t < TSEQ; ++t) {
        const int pw = t & 1, pr = pw ^ 1;
        const float* xt  = x + (size_t)t * 15;
        // ---- layer 0: gates = Wih0 @ x_t + Whh0 @ h0_prev + b ----
        {
            const float* hr = h0 + pr * HID;
            float acc[4];
            #pragma unroll
            for (int g = 0; g < 4; ++g) {
                const int row = g * HID + j;
                const float* wr = Whh0 + (size_t)row * HID;
                float s = 0.0f;
                for (int k = lane; k < HID; k += 32) s += wr[k] * hr[k];
                if (lane < 15) s += Wih0[row * 15 + lane] * xt[lane];
                acc[g] = wave_reduce(s);
            }
            if (lane == 0) {
                float gi = acc[0] + bih0[j]           + bhh0[j];
                float gf = acc[1] + bih0[HID + j]     + bhh0[HID + j];
                float gg = acc[2] + bih0[2 * HID + j] + bhh0[2 * HID + j];
                float go = acc[3] + bih0[3 * HID + j] + bhh0[3 * HID + j];
                float cn = sigm(gf) * c0[j] + sigm(gi) * tanhf(gg);
                c0[j] = cn;
                h0[pw * HID + j] = sigm(go) * tanhf(cn);
            }
        }
        grid_sync(cnt, phase, lp, NBLK);
        // ---- layer 1: gates = Wih1 @ h0_new + Whh1 @ h1_prev + b ----
        {
            const float* in1 = h0 + pw * HID;
            const float* hr  = h1 + pr * HID;
            float acc[4];
            #pragma unroll
            for (int g = 0; g < 4; ++g) {
                const int row = g * HID + j;
                const float* wi = Wih1 + (size_t)row * HID;
                const float* wh = Whh1 + (size_t)row * HID;
                float s = 0.0f;
                for (int k = lane; k < HID; k += 32) s += wi[k] * in1[k] + wh[k] * hr[k];
                acc[g] = wave_reduce(s);
            }
            if (lane == 0) {
                float gi = acc[0] + bih1[j]           + bhh1[j];
                float gf = acc[1] + bih1[HID + j]     + bhh1[HID + j];
                float gg = acc[2] + bih1[2 * HID + j] + bhh1[2 * HID + j];
                float go = acc[3] + bih1[3 * HID + j] + bhh1[3 * HID + j];
                float cn = sigm(gf) * c1[j] + sigm(gi) * tanhf(gg);
                c1[j] = cn;
                float hn = sigm(go) * tanhf(cn);
                h1[pw * HID + j] = hn;
                hs[(size_t)t * HID + j] = f32_to_bf16(hn);
            }
        }
        grid_sync(cnt, phase, lp, NBLK);
    }
}

// ---------------------------------------------------------------------------
// Y = leaky(X @ W^T + bias), X:[M,1024] bf16, W:[1024,1024] bf16 row-major.
// TDM stages a 128x64 A tile + 16x64 B tile into LDS per K-chunk; 8 waves
// consume them via LDS b128 reads feeding v_wmma_f32_16x16x32_bf16.
// Grid (M/128, 64), block 256.
__global__ void mlp_gemm_wmma(const unsigned short* __restrict__ X,
                              const unsigned short* __restrict__ W,
                              const float* __restrict__ bias,
                              unsigned short* __restrict__ Y, int M) {
    __shared__ unsigned short smA[128 * 64];   // 16 KB
    __shared__ unsigned short smB[16 * 64];    //  2 KB
    const int lane = threadIdx.x & 31;
    const int wid  = threadIdx.x >> 5;
    const int mblk = blockIdx.x * 128;
    const int m0 = mblk + wid * 16;
    const int n0 = blockIdx.y * 16;
    const int l15 = lane & 15;
    const int hi  = lane >> 4;                 // 0: lanes 0-15, 1: lanes 16-31
    const int ka   = hi * 8;   // A frag: K halves {0..7,16..23} vs {8..15,24..31}
    const int kb16 = hi * 16;  // unused: B frag uses ka, same K halves as A

    // LDS byte offsets (flat-aperture rule: LDS offset is the low 32 bits of the generic address).
    const unsigned ldsA = (unsigned)(uintptr_t)(void*)smA;
    const unsigned ldsB = (unsigned)(uintptr_t)(void*)smB;

    const unsigned short* arow = smA + (wid * 16 + l15) * 64;  // A row in tile
    const unsigned short* brow = smB + l15 * 64;               // B row in tile

    v8f acc = {};
    for (int kb = 0; kb < HID; kb += 64) {
        if (threadIdx.x == 0) {
            // A tile: rows mblk..mblk+127, cols kb..kb+63 of X [M,1024]
            tdm_load_tile_2d(X + (size_t)mblk * HID + kb, ldsA,
                             HID, (unsigned)M, HID, 64, 128);
            // B tile: rows n0..n0+15, cols kb..kb+63 of W [1024,1024]
            tdm_load_tile_2d(W + (size_t)n0 * HID + kb, ldsB,
                             HID, HID, HID, 64, 16);
            __builtin_amdgcn_s_wait_tensorcnt(0);
        }
        __syncthreads();
        #pragma unroll
        for (int kk = 0; kk < 64; kk += 32) {
            v8bf alo = *(const v8bf*)(arow + kk + ka);
            v8bf ahi = *(const v8bf*)(arow + kk + 16 + ka);
            v8bf blo = *(const v8bf*)(brow + kk + ka);
            v8bf bhi = *(const v8bf*)(brow + kk + 16 + ka);
            v16bf a, b;
            #pragma unroll
            for (int i = 0; i < 8; ++i) {
                a[i] = alo[i]; a[8 + i] = ahi[i];
                b[i] = blo[i]; b[8 + i] = bhi[i];
            }
            acc = __builtin_amdgcn_wmma_f32_16x16x32_bf16(
                /*neg_a=*/false, a, /*neg_b=*/false, b,
                /*c_mod=*/(short)0, acc, /*reuse_a=*/false, /*reuse_b=*/false);
        }
        __syncthreads();
    }
    // C/D layout: VGPR v holds M = m0 + v (+8 for lanes 16-31), N = n0 + (lane&15)
    const int nn = n0 + l15;
    const int mb = m0 + hi * 8;
    const float bv = bias[nn];
    #pragma unroll
    for (int v = 0; v < 8; ++v) {
        float y = leaky(acc[v] + bv);
        Y[(size_t)(mb + v) * HID + nn] = f32_to_bf16(y);
    }
}

// ---------------------------------------------------------------------------
// out[t, :] = leaky(X4[t] @ Wo^T + bo). One wave per row t.
__global__ void out_head(const unsigned short* __restrict__ X4,
                         const float* __restrict__ Wo, const float* __restrict__ bo,
                         float* __restrict__ out, int M) {
    const int lane = threadIdx.x & 31;
    const int w = (blockIdx.x * blockDim.x + threadIdx.x) >> 5;
    if (w >= M) return;
    const unsigned short* xr = X4 + (size_t)w * HID;
    for (int o = 0; o < 15; ++o) {
        const float* wr = Wo + o * HID;
        float s = 0.0f;
        for (int k = lane; k < HID; k += 32) s += bf16_to_f32(xr[k]) * wr[k];
        s = wave_reduce(s);
        if (lane == 0) out[(size_t)w * 15 + o] = leaky(s + bo[o]);
    }
}

// ---------------------------------------------------------------------------
// 16 closed-loop future steps: LSTM(2 layers) + 5-layer MLP, all mat-vec,
// persistent with grid barriers. One wave per output unit.
__global__ void future_kernel(
    const float* __restrict__ Wih0, const float* __restrict__ Whh0,
    const float* __restrict__ bih0, const float* __restrict__ bhh0,
    const float* __restrict__ Wih1, const float* __restrict__ Whh1,
    const float* __restrict__ bih1, const float* __restrict__ bhh1,
    const float* __restrict__ linW, const float* __restrict__ linb,
    const float* __restrict__ W1, const float* __restrict__ b1,
    const float* __restrict__ W2, const float* __restrict__ b2,
    const float* __restrict__ Wo, const float* __restrict__ bo,
    float* __restrict__ h0, float* __restrict__ c0,
    float* __restrict__ h1, float* __restrict__ c1,
    float* __restrict__ vA, float* __restrict__ vB,
    float* __restrict__ outp, unsigned* __restrict__ bar) {
    const int lane = threadIdx.x & 31;
    const int wid  = threadIdx.x >> 5;
    const int j    = blockIdx.x * 8 + wid;
    unsigned lp = 0;
    unsigned* cnt = bar; unsigned* phase = bar + 1;
    int rp = (TSEQ - 1) & 1;   // parity holding the final scan state

    for (int it = 0; it < NFUT; ++it) {
        const int wp = rp ^ 1;
        const float* inrow = outp + (size_t)(TSEQ - 1 + it) * 15;
        // layer 0
        {
            const float* hr = h0 + rp * HID;
            float acc[4];
            #pragma unroll
            for (int g = 0; g < 4; ++g) {
                const int row = g * HID + j;
                const float* wr = Whh0 + (size_t)row * HID;
                float s = 0.0f;
                for (int k = lane; k < HID; k += 32) s += wr[k] * hr[k];
                if (lane < 15) s += Wih0[row * 15 + lane] * inrow[lane];
                acc[g] = wave_reduce(s);
            }
            if (lane == 0) {
                float gi = acc[0] + bih0[j]           + bhh0[j];
                float gf = acc[1] + bih0[HID + j]     + bhh0[HID + j];
                float gg = acc[2] + bih0[2 * HID + j] + bhh0[2 * HID + j];
                float go = acc[3] + bih0[3 * HID + j] + bhh0[3 * HID + j];
                float cn = sigm(gf) * c0[j] + sigm(gi) * tanhf(gg);
                c0[j] = cn;
                h0[wp * HID + j] = sigm(go) * tanhf(cn);
            }
        }
        grid_sync(cnt, phase, lp, NBLK);
        // layer 1
        {
            const float* in1 = h0 + wp * HID;
            const float* hr  = h1 + rp * HID;
            float acc[4];
            #pragma unroll
            for (int g = 0; g < 4; ++g) {
                const int row = g * HID + j;
                const float* wi = Wih1 + (size_t)row * HID;
                const float* wh = Whh1 + (size_t)row * HID;
                float s = 0.0f;
                for (int k = lane; k < HID; k += 32) s += wi[k] * in1[k] + wh[k] * hr[k];
                acc[g] = wave_reduce(s);
            }
            if (lane == 0) {
                float gi = acc[0] + bih1[j]           + bhh1[j];
                float gf = acc[1] + bih1[HID + j]     + bhh1[HID + j];
                float gg = acc[2] + bih1[2 * HID + j] + bhh1[2 * HID + j];
                float go = acc[3] + bih1[3 * HID + j] + bhh1[3 * HID + j];
                float cn = sigm(gf) * c1[j] + sigm(gi) * tanhf(gg);
                c1[j] = cn;
                float hn = sigm(go) * tanhf(cn);
                h1[wp * HID + j] = hn;
                vA[j] = hn;
            }
        }
        grid_sync(cnt, phase, lp, NBLK);
        // 4 hidden MLP layers: vA -> vB -> vA -> vB -> vA
        #pragma unroll 1
        for (int L = 0; L < 4; ++L) {
            const float* Wl = (L == 0) ? linW : (L == 1) ? (linW + HID * HID)
                              : (L == 2) ? W1 : W2;
            const float* bl = (L == 0) ? linb : (L == 1) ? (linb + HID)
                              : (L == 2) ? b1 : b2;
            const float* src = (L & 1) ? vB : vA;
            float*       dst = (L & 1) ? vA : vB;
            const float* wr = Wl + (size_t)j * HID;
            float s = 0.0f;
            for (int k = lane; k < HID; k += 32) s += wr[k] * src[k];
            s = wave_reduce(s);
            if (lane == 0) dst[j] = leaky(s + bl[j]);
            grid_sync(cnt, phase, lp, NBLK);
        }
        // output head (15 units) from vA
        if (j < 15) {
            const float* wr = Wo + (size_t)j * HID;
            float s = 0.0f;
            for (int k = lane; k < HID; k += 32) s += wr[k] * vA[k];
            s = wave_reduce(s);
            if (lane == 0) outp[(size_t)(TSEQ + it) * 15 + j] = leaky(s + bo[j]);
        }
        grid_sync(cnt, phase, lp, NBLK);
        rp = wp;
    }
}

// ---------------------------------------------------------------------------
extern "C" void kernel_launch(void* const* d_in, const int* in_sizes, int n_in,
                              void* d_out, int out_size, void* d_ws, size_t ws_size,
                              hipStream_t stream) {
    (void)in_sizes; (void)n_in; (void)out_size; (void)ws_size;
    const float* x    = (const float*)d_in[0];
    const float* Wih0 = (const float*)d_in[1];
    const float* Whh0 = (const float*)d_in[2];
    const float* bih0 = (const float*)d_in[3];
    const float* bhh0 = (const float*)d_in[4];
    const float* Wih1 = (const float*)d_in[5];
    const float* Whh1 = (const float*)d_in[6];
    const float* bih1 = (const float*)d_in[7];
    const float* bhh1 = (const float*)d_in[8];
    const float* linW = (const float*)d_in[9];
    const float* linb = (const float*)d_in[10];
    const float* W1   = (const float*)d_in[11];
    const float* b1   = (const float*)d_in[12];
    const float* W2   = (const float*)d_in[13];
    const float* b2   = (const float*)d_in[14];
    const float* Wo   = (const float*)d_in[15];
    const float* bo   = (const float*)d_in[16];
    float* out = (float*)d_out;

    char* wsb = (char*)d_ws;
    unsigned* bar_scan = (unsigned*)(wsb);
    unsigned* bar_fut  = (unsigned*)(wsb + 64);
    float* h0 = (float*)(wsb + 128);
    float* h1 = h0 + 2 * HID;
    float* c0 = h1 + 2 * HID;
    float* c1 = c0 + HID;
    float* vA = c1 + HID;
    float* vB = vA + HID;
    unsigned short* Wb = (unsigned short*)(wsb + 33024);   // 4 x H*H bf16 (8 MB)
    unsigned short* hs = Wb + 4u * HID * HID;              // [T,H] bf16 (16 MB)
    unsigned short* xb = hs + (size_t)TSEQ * HID;          // ping buffer (16 MB)

    // Zero barriers + h/c state (+ vec buffers) each launch -> deterministic.
    (void)hipMemsetAsync(d_ws, 0, 33024, stream);

    convert_w<<<4096, 256, 0, stream>>>(linW, linW + HID * HID, W1, W2, Wb);

    lstm_scan_kernel<<<NBLK, 256, 0, stream>>>(x, Wih0, Whh0, bih0, bhh0,
                                               Wih1, Whh1, bih1, bhh1,
                                               h0, c0, h1, c1, hs, bar_scan);

    dim3 gg(TSEQ / 128, HID / 16);   // (64, 64)
    mlp_gemm_wmma<<<gg, 256, 0, stream>>>(hs, Wb,                  linb,        xb, TSEQ);
    mlp_gemm_wmma<<<gg, 256, 0, stream>>>(xb, Wb + 1 * HID * HID,  linb + HID,  hs, TSEQ);
    mlp_gemm_wmma<<<gg, 256, 0, stream>>>(hs, Wb + 2 * HID * HID,  b1,          xb, TSEQ);
    mlp_gemm_wmma<<<gg, 256, 0, stream>>>(xb, Wb + 3 * HID * HID,  b2,          hs, TSEQ);

    out_head<<<1024, 256, 0, stream>>>(hs, Wo, bo, out, TSEQ);

    future_kernel<<<NBLK, 256, 0, stream>>>(Wih0, Whh0, bih0, bhh0,
                                            Wih1, Whh1, bih1, bhh1,
                                            linW, linb, W1, b1, W2, b2, Wo, bo,
                                            h0, c0, h1, c1, vA, vB, out, bar_fut);
}
